// VQ_66881230733865
// MI455X (gfx1250) — matrix operand order repaired and checked by reading, not compiled, not run on hardware
//
#include <hip/hip_runtime.h>

typedef __attribute__((ext_vector_type(16))) __bf16 v16bf;
typedef __attribute__((ext_vector_type(8)))  __bf16 v8bf;
typedef __attribute__((ext_vector_type(8)))  float  v8f;
typedef __attribute__((ext_vector_type(4)))  float  v4f;
typedef __attribute__((ext_vector_type(4)))  unsigned int u32x4;
typedef __attribute__((ext_vector_type(8)))  int          i32x8;
typedef __attribute__((ext_vector_type(4)))  int          i32x4;

#define K_EMB 1024
#define DDIM  256
#define HW    4096
#define NIMG  32
#define NPIX  (NIMG * HW)   // 131072 pixels

// ---------------------------------------------------------------------------
// Prep: emb fp32 [1024,256] -> bf16 row-major in d_ws, plus b2[k] = ||emb_k||^2
// ---------------------------------------------------------------------------
__global__ void vq_prep_kernel(const float* __restrict__ emb,
                               unsigned short* __restrict__ ebf_raw,
                               float* __restrict__ b2) {
  __bf16* ebf = reinterpret_cast<__bf16*>(ebf_raw);
  const int c = blockIdx.x;    // code
  const int t = threadIdx.x;   // d
  const float w = emb[c * DDIM + t];
  ebf[c * DDIM + t] = (__bf16)w;
  __shared__ float red[DDIM];
  red[t] = w * w;
  __syncthreads();
  for (int s = DDIM / 2; s > 0; s >>= 1) {
    if (t < s) red[t] += red[t + s];
    __syncthreads();
  }
  if (t == 0) b2[c] = red[0];
}

// ---------------------------------------------------------------------------
// TDM: DMA one 16x256 bf16 codebook tile (8 KB) from global into LDS.
// D# built per CDNA5 ISA 8.3/8.4: 2D tensor [1024 x 256] bf16 row-major,
// tile = 16 rows x 256 cols, data_size=2B, type=2 ("image").
// Toolchain here exposes the 6-arg builtin:
//   (u32x4 g0, i32x8 g1, i32x4 g2, i32x4 g3, i32x8 extra, i32 cpol)
// ---------------------------------------------------------------------------
__device__ __forceinline__ void tdm_load_codes(const unsigned short* gsrc,
                                               unsigned int lds_addr) {
  const unsigned long long ga = (unsigned long long)(uintptr_t)gsrc;
  u32x4 g0;
  g0[0] = 1u;                                   // count=1 (valid user D#)
  g0[1] = lds_addr;                             // LDS byte address
  g0[2] = (unsigned int)(ga & 0xffffffffu);     // global_addr[31:0]
  g0[3] = (unsigned int)((ga >> 32) & 0x01ffffffu) | (2u << 30); // [56:32]|type=2
  i32x8 g1;
  g1[0] = 0x00010000;            // wg_mask=0, data_size=1 (2 bytes)
  g1[1] = (int)(256u << 16);     // tensor_dim0 = 256 (low 16 in [31:16])
  g1[2] = (int)(1024u << 16);    // tensor_dim0 hi=0 | tensor_dim1 = 1024
  g1[3] = (int)(256u << 16);     // tensor_dim1 hi=0 | tile_dim0 = 256
  g1[4] = 16;                    // tile_dim1 = 16, tile_dim2 = 0
  g1[5] = 256;                   // tensor_dim0_stride = 256 (low 32)
  g1[6] = 0;                     // stride hi / tensor_dim1_stride lo
  g1[7] = 0;
  i32x4 g2 = {0, 0, 0, 0};
  i32x4 g3 = {0, 0, 0, 0};
  i32x8 gx = {0, 0, 0, 0, 0, 0, 0, 0};
  __builtin_amdgcn_tensor_load_to_lds(g0, g1, g2, g3, gx, 0);
}

// ---------------------------------------------------------------------------
// Main: each wave32 handles 32 pixels (2 N-tiles) vs all 1024 codes.
//   A (SRC0) = 16x32 bf16 codebook tile, staged in LDS by the TDM,
//              shared by all 8 waves of the block (ds_load_b128 reads).
//   B (SRC1) = 32x16 bf16 z tile per N-tile (lane = pixel), held in VGPRs.
//   C/D      = 16x16 f32 (row = code, col = pixel); 2 independent acc chains.
// dist = b2[code] - 2*ab  (||z||^2 per-pixel constant -> dropped for argmin)
// ---------------------------------------------------------------------------
__global__ void __launch_bounds__(256, 1) vq_main_kernel(
    const float* __restrict__ z, const float* __restrict__ emb,
    const unsigned short* __restrict__ ebf_raw,
    const float* __restrict__ b2, float* __restrict__ out) {
  __shared__ __align__(256) unsigned short lds_codes[2][16 * DDIM]; // 2 x 8 KB

  const int lane = threadIdx.x & 31;
  const int wvb  = threadIdx.x >> 5;          // wave in block (0..7)
  const int p    = lane & 15;                 // pixel-in-tile / A code row
  const int half = lane >> 4;                 // lane half
  const int pixbase = blockIdx.x * 256 + wvb * 32;

  size_t zb[2];
#pragma unroll
  for (int t = 0; t < 2; ++t) {
    const int g = pixbase + t * 16 + p;
    zb[t] = (size_t)(g >> 12) * DDIM * HW + (size_t)(g & (HW - 1));
  }

  // --- z fragments for both N-tiles (B matrix), cached in VGPRs ---
  v16bf bfrag[2][8];
#pragma unroll
  for (int t = 0; t < 2; ++t)
#pragma unroll
    for (int ch = 0; ch < 8; ++ch) {
      const int dbase = ch * 32 + half * 8;
#pragma unroll
      for (int j = 0; j < 16; ++j)
        bfrag[t][ch][j] = (__bf16)z[zb[t] + (size_t)(dbase + j + (j >= 8 ? 8 : 0)) * HW];
    }

  float minv[2] = {3.0e38f, 3.0e38f};
  int   mini[2] = {0, 0};

  // Prologue: wave 0 DMAs k-tile 0 into buffer 0.
  if (threadIdx.x < 32)
    tdm_load_codes(ebf_raw, (unsigned int)(uintptr_t)&lds_codes[0][0]);
  __builtin_amdgcn_s_wait_tensorcnt(0);
  __syncthreads();

  for (int kt = 0; kt < 64; ++kt) {
    const int cur = kt & 1;
    // Overlap: DMA next k-tile into the other buffer while we compute.
    if ((kt + 1) < 64 && threadIdx.x < 32)
      tdm_load_codes(ebf_raw + (size_t)(kt + 1) * 16 * DDIM,
                     (unsigned int)(uintptr_t)&lds_codes[cur ^ 1][0]);

    v8f acc0 = {}, acc1 = {};
    const __bf16* lbase = reinterpret_cast<const __bf16*>(&lds_codes[cur][0]);
#pragma unroll
    for (int ch = 0; ch < 8; ++ch) {
      // A layout: lanes<16 -> K in {d0..d0+7, d0+16..d0+23}; lanes>=16 -> +8.
      const __bf16* ls = lbase + p * DDIM + ch * 32 + half * 8;
      const v8bf lo = *(const v8bf*)ls;          // ds_load_b128
      const v8bf hi = *(const v8bf*)(ls + 16);   // ds_load_b128
      v16bf af;
#pragma unroll
      for (int j = 0; j < 8; ++j) { af[j] = lo[j]; af[8 + j] = hi[j]; }
      // Two independent accumulator chains -> back-to-back XDL issue.
      acc0 = __builtin_amdgcn_wmma_f32_16x16x32_bf16(
          false, af, false, bfrag[0][ch], (short)0, acc0, false, false);
      acc1 = __builtin_amdgcn_wmma_f32_16x16x32_bf16(
          false, af, false, bfrag[1][ch], (short)0, acc1, false, false);
    }

    // C layout: VGPR v = row (v + 8*half), col = lane&15 (= pixel p).
    const float* bp = b2 + kt * 16 + half * 8;
    const v4f b2a = *(const v4f*)bp;
    const v4f b2b = *(const v4f*)(bp + 4);
#pragma unroll
    for (int v = 0; v < 8; ++v) {
      const float bb = (v < 4) ? b2a[v] : b2b[v - 4];
      const int ci = kt * 16 + half * 8 + v;
      const float d0 = bb - 2.0f * acc0[v];
      if (d0 < minv[0]) { minv[0] = d0; mini[0] = ci; }
      const float d1 = bb - 2.0f * acc1[v];
      if (d1 < minv[1]) { minv[1] = d1; mini[1] = ci; }
    }

    __builtin_amdgcn_s_wait_tensorcnt(0);  // next buffer landed
    __syncthreads();                       // everyone done reading cur buffer
  }

  const size_t STOFF = (size_t)NIMG * DDIM * HW;  // 33554432
  const size_t IOFF  = 2 * STOFF;                 // 67108864

#pragma unroll
  for (int t = 0; t < 2; ++t) {
    // Combine the two lane-halves (codes v vs v+8 of each tile, same pixel).
    float mv = minv[t];
    int   mi = mini[t];
    const float om = __shfl_xor(mv, 16, 32);
    const int   oi = __shfl_xor(mi, 16, 32);
    if (om < mv || (om == mv && oi < mi)) { mv = om; mi = oi; }

    const int g = pixbase + t * 16 + p;
    if (half == 0) out[IOFF + g] = (float)mi;     // indxs [n,h,w] as float

    // Gather emb[mi][:] back to NCHW; each lane-half writes 128 channels.
    const float* qr = emb + (size_t)mi * DDIM;
#pragma unroll 4
    for (int j = 0; j < 128; ++j) {
      const int d = half * 128 + j;
      const float qv = qr[d];
      const size_t o = zb[t] + (size_t)d * HW;    // n*D*HW + d*HW + hw
      const float zz = z[o];
      out[o] = qv;                                // quantized
      out[STOFF + o] = zz + (qv - zz);            // straight-through
    }
  }
}

// ---------------------------------------------------------------------------
extern "C" void kernel_launch(void* const* d_in, const int* in_sizes, int n_in,
                              void* d_out, int out_size, void* d_ws, size_t ws_size,
                              hipStream_t stream) {
  (void)in_sizes; (void)n_in; (void)out_size; (void)ws_size;
  const float* z   = (const float*)d_in[0];      // [32,256,64,64]
  const float* emb = (const float*)d_in[1];      // [1024,256]

  unsigned short* ebf = (unsigned short*)d_ws;                    // 512 KB bf16 codebook
  float* b2 = (float*)((char*)d_ws + (size_t)K_EMB * DDIM * 2);   // 4 KB norms

  vq_prep_kernel<<<K_EMB, DDIM, 0, stream>>>(emb, ebf, b2);
  // 131072 pixels / (8 waves * 32 pixels) = 512 blocks of 256 threads
  vq_main_kernel<<<NPIX / 256, 256, 0, stream>>>(z, emb, ebf, b2, (float*)d_out);
}
